// Encoder_17231408791768
// MI455X (gfx1250) — compile-verified, matrix operand order checked by reading, not run on hardware
//
#include <hip/hip_runtime.h>
#include <hip/hip_bf16.h>
#include <stddef.h>

// ---------------------------------------------------------------------------
// GraphConv x3 for MI455X (gfx1250, wave32, WMMA).
//   per layer: agg = scatter_add(x[src]*ew -> dst)   [f32 atomics, L2-resident]
//              out = agg @ w_rel^T + x @ w_root^T + b [+ReLU]  via
//              v_wmma_f32_16x16x32_bf16 (bf16 in, f32 accumulate)
// ---------------------------------------------------------------------------

#define N_NODES 50000
#define N_EDGES 800000
#define DFEAT   128

typedef __attribute__((ext_vector_type(16))) __bf16 v16bf;
typedef __attribute__((ext_vector_type(8)))  float  v8f;

// K-pair base for 16-bit A/B fragment layout (ISA 7.12.2):
// lanes 0-15 hold K = {0..7, 16..23} as pairs in VGPRs 0..7,
// lanes 16-31 hold K = {8..15, 24..31}.
__device__ __forceinline__ int kbase_of(int p, int half) {
    return (p < 4) ? (p * 2 + half * 8) : (16 + (p - 4) * 2 + half * 8);
}

// ---------------------------------------------------------------------------
// Zero the aggregation buffer.
__global__ void zero_kernel(float* __restrict__ p, int n) {
    int i = blockIdx.x * blockDim.x + threadIdx.x;
    if (i < n) p[i] = 0.0f;
}

// ---------------------------------------------------------------------------
// Convert + swizzle six 128x128 f32 weight matrices into bf16 B-fragment
// order: [mat][kt(4)][ct(8)][lane(32)][16 bf16] -> each wave lane later loads
// one contiguous 32-byte chunk per (kt,ct) fragment.
__global__ void wconv_kernel(const float* __restrict__ w0, const float* __restrict__ w1,
                             const float* __restrict__ w2, const float* __restrict__ w3,
                             const float* __restrict__ w4, const float* __restrict__ w5,
                             __bf16* __restrict__ wbf) {
    int gid = blockIdx.x * blockDim.x + threadIdx.x;
    if (gid >= 6 * 16384) return;
    int mat  = gid >> 14;
    int r    = gid & 16383;
    int i    = r & 15;          // element within lane's 16
    int lane = (r >> 4) & 31;
    int ct   = (r >> 9) & 7;    // output-column tile
    int kt   = (r >> 12) & 3;   // K tile (32 per tile)
    int half = lane >> 4;
    int n    = lane & 15;
    int o    = ct * 16 + n;                 // output feature (weight row)
    int p    = i >> 1;
    int lo   = i & 1;
    int k    = kt * 32 + kbase_of(p, half) + lo;
    const float* srcs[6] = {w0, w1, w2, w3, w4, w5};
    wbf[gid] = (__bf16)srcs[mat][o * DFEAT + k];
}

// ---------------------------------------------------------------------------
// Edge scatter: one wave (32 lanes) per edge; lane handles 4 contiguous
// features (float4 load = fully coalesced 512B per wave), scales by edge
// weight, atomic-adds into agg[dst]. f32 atomics resolve in L2 atomic units.
__global__ void scatter_kernel(const float* __restrict__ x,
                               const int* __restrict__ src,
                               const int* __restrict__ dst,
                               const float* __restrict__ ew,
                               float* __restrict__ agg, int nedges) {
    int t = blockIdx.x * blockDim.x + threadIdx.x;
    int e = t >> 5;
    if (e >= nedges) return;
    int lane = t & 31;
    int s = src[e];
    int d = dst[e];
    float w = ew[e];
    const float4 v = *reinterpret_cast<const float4*>(x + (size_t)s * DFEAT + lane * 4);
    float* o = agg + (size_t)d * DFEAT + lane * 4;
    atomicAdd(o + 0, v.x * w);
    atomicAdd(o + 1, v.y * w);
    atomicAdd(o + 2, v.z * w);
    atomicAdd(o + 3, v.w * w);
}

// ---------------------------------------------------------------------------
// One K-sweep (128 = 4 x 32) of one activation matrix against one pre-swizzled
// weight matrix, accumulating into 8 column-tile accumulators.
__device__ __forceinline__ void gemm_pass(v8f acc[8],
                                          const float* __restrict__ act,
                                          const __bf16* __restrict__ wm,
                                          int r0, int lane) {
    const int half = lane >> 4;
    const int m    = lane & 15;
    const float* arow = act + (size_t)(r0 + m) * DFEAT;
#pragma unroll
    for (int kt = 0; kt < 4; ++kt) {
        // Build A fragment (16x32 bf16) from f32 activations, ISA lane layout.
        v16bf a;
#pragma unroll
        for (int p = 0; p < 8; ++p) {
            int kb = kbase_of(p, half);
            float2 f = *reinterpret_cast<const float2*>(arow + kt * 32 + kb);
            a[2 * p]     = (__bf16)f.x;
            a[2 * p + 1] = (__bf16)f.y;
        }
#pragma unroll
        for (int ct = 0; ct < 8; ++ct) {
            v16bf b = *reinterpret_cast<const v16bf*>(
                wm + ((size_t)((kt * 8 + ct) * 32 + lane)) * 16);
            acc[ct] = __builtin_amdgcn_wmma_f32_16x16x32_bf16(
                false, a, false, b, (short)0, acc[ct], false, false);
        }
    }
}

// ---------------------------------------------------------------------------
// out[16 rows x 128] per wave: agg @ w_rel^T + x @ w_root^T + b (+ReLU).
// 4 waves / block; wave-uniform early exit keeps EXEC all-ones for WMMA.
__global__ void gemm_kernel(const float* __restrict__ agg,
                            const float* __restrict__ xin,
                            const __bf16* __restrict__ wrel,
                            const __bf16* __restrict__ wroot,
                            const float* __restrict__ bias,
                            float* __restrict__ out,
                            int nrows, int do_relu) {
    const int lane = threadIdx.x & 31;
    const int wave = threadIdx.x >> 5;
    const int tile = blockIdx.x * 4 + wave;
    if (tile * 16 >= nrows) return;           // whole-wave uniform
    const int r0   = tile * 16;
    const int half = lane >> 4;
    const int n    = lane & 15;

    v8f acc[8] = {};
    gemm_pass(acc, agg, wrel, r0, lane);      // neighbor aggregate @ w_rel^T
    gemm_pass(acc, xin, wroot, r0, lane);     // root features     @ w_root^T

    // Epilogue: C/D layout — VGPR j holds row (j + 8*half), lane&15 is column.
#pragma unroll
    for (int ct = 0; ct < 8; ++ct) {
        int col = ct * 16 + n;
        float bv = bias[col];
#pragma unroll
        for (int j = 0; j < 8; ++j) {
            int row = r0 + j + half * 8;
            float v = acc[ct][j] + bv;
            if (do_relu) v = fmaxf(v, 0.0f);
            out[(size_t)row * DFEAT + col] = v;
        }
    }
}

// ---------------------------------------------------------------------------
extern "C" void kernel_launch(void* const* d_in, const int* in_sizes, int n_in,
                              void* d_out, int out_size, void* d_ws, size_t ws_size,
                              hipStream_t stream) {
    (void)in_sizes; (void)n_in; (void)out_size; (void)ws_size;

    const float* x    = (const float*)d_in[0];
    const int*   eidx = (const int*)d_in[1];
    const float* ew   = (const float*)d_in[2];
    const float* w1r  = (const float*)d_in[3];
    const float* w1o  = (const float*)d_in[4];
    const float* b1   = (const float*)d_in[5];
    const float* w2r  = (const float*)d_in[6];
    const float* w2o  = (const float*)d_in[7];
    const float* b2   = (const float*)d_in[8];
    const float* w3r  = (const float*)d_in[9];
    const float* w3o  = (const float*)d_in[10];
    const float* b3   = (const float*)d_in[11];

    const int* src = eidx;            // edge_index[0]
    const int* dst = eidx + N_EDGES;  // edge_index[1]

    const size_t feat_elems = (size_t)N_NODES * DFEAT;           // 6.4M floats
    const size_t feat_bytes = feat_elems * sizeof(float);        // 25.6 MB (256B-aligned)
    char* ws = (char*)d_ws;
    float*  agg = (float*)ws;
    float*  h   = (float*)(ws + feat_bytes);
    __bf16* wbf = (__bf16*)(ws + 2 * feat_bytes);                // 6 * 16384 bf16

    const int ZB = ((int)feat_elems + 255) / 256;                // zero blocks
    const int SB = (N_EDGES * 32 + 255) / 256;                   // scatter blocks
    const int GB = ((N_NODES / 16) + 3) / 4;                     // gemm blocks (4 waves ea.)

    // One-time weight convert + swizzle into bf16 fragment order.
    wconv_kernel<<<(6 * 16384 + 255) / 256, 256, 0, stream>>>(
        w1r, w1o, w2r, w2o, w3r, w3o, wbf);

    // ---- Layer 1: x -> h (ReLU) ----
    zero_kernel<<<ZB, 256, 0, stream>>>(agg, (int)feat_elems);
    scatter_kernel<<<SB, 256, 0, stream>>>(x, src, dst, ew, agg, N_EDGES);
    gemm_kernel<<<GB, 128, 0, stream>>>(agg, x, wbf + 0 * 16384, wbf + 1 * 16384,
                                        b1, h, N_NODES, 1);

    // ---- Layer 2: h -> h in place (ReLU) ----
    zero_kernel<<<ZB, 256, 0, stream>>>(agg, (int)feat_elems);
    scatter_kernel<<<SB, 256, 0, stream>>>(h, src, dst, ew, agg, N_EDGES);
    gemm_kernel<<<GB, 128, 0, stream>>>(agg, h, wbf + 2 * 16384, wbf + 3 * 16384,
                                        b2, h, N_NODES, 1);

    // ---- Layer 3: h -> out (no ReLU) ----
    zero_kernel<<<ZB, 256, 0, stream>>>(agg, (int)feat_elems);
    scatter_kernel<<<SB, 256, 0, stream>>>(h, src, dst, ew, agg, N_EDGES);
    gemm_kernel<<<GB, 128, 0, stream>>>(agg, h, wbf + 4 * 16384, wbf + 5 * 16384,
                                        b3, (float*)d_out, N_NODES, 0);
}
